// GINNet_30210799960807
// MI455X (gfx1250) — matrix operand order repaired and checked by reading, not compiled, not run on hardware
//
#include <hip/hip_runtime.h>
#include <hip/hip_bf16.h>

typedef __attribute__((ext_vector_type(16))) _Float16 v16h;
typedef __attribute__((ext_vector_type(8)))  float    v8f;

#define BN_EPS 1e-5f

static __device__ inline v8f wmma_f16(v16h a, v16h b, v8f c) {
    // (neg_a, A, neg_b, B, c_mod, C, reuse_a, reuse_b)
    return __builtin_amdgcn_wmma_f32_16x16x32_f16(false, a, false, b, (short)0, c,
                                                  false, false);
}

static __device__ inline v8f zero8() {
    v8f z = {0.f, 0.f, 0.f, 0.f, 0.f, 0.f, 0.f, 0.f};
    return z;
}

// ---------------------------------------------------------------------------
// float4 copy: dst = src (n4 = number of float4)
// ---------------------------------------------------------------------------
__global__ void copy_f4_kernel(const float* __restrict__ src, float* __restrict__ dst,
                               long n4) {
    long tid = (long)blockIdx.x * blockDim.x + threadIdx.x;
    if (tid < n4) {
        reinterpret_cast<float4*>(dst)[tid] =
            reinterpret_cast<const float4*>(src)[tid];
    }
}

// ---------------------------------------------------------------------------
// Edge aggregation: agg[dst] += x[src] for all edges, F features, float4 chunks
// ---------------------------------------------------------------------------
template <int F>
__global__ void edge_aggr_kernel(const float* __restrict__ xin,
                                 const int* __restrict__ src,
                                 const int* __restrict__ dst,
                                 float* __restrict__ agg, int n_edges) {
    const int vecs = F / 4;
    long tid = (long)blockIdx.x * blockDim.x + threadIdx.x;
    long total = (long)n_edges * vecs;
    if (tid >= total) return;
    int e = (int)(tid / vecs);
    int c = (int)(tid % vecs) * 4;
    int s = src[e];
    int d = dst[e];
    float4 v = *reinterpret_cast<const float4*>(xin + (long)s * F + c);
    float* p = agg + (long)d * F + c;
    atomicAdd(p + 0, v.x);
    atomicAdd(p + 1, v.y);
    atomicAdd(p + 2, v.z);
    atomicAdd(p + 3, v.w);
}

// ---------------------------------------------------------------------------
// Fused GIN MLP: r = relu( relu(agg@wa + ba) @ wb + bb ), accumulating BN
// sum / sumsq into stats[0..31] / stats[32..63].
// Block = 128 threads = 4 waves; each wave owns one 16-node tile.
// WMMA f32_16x16x32_f16, K_IN in {64, 32}, DIM fixed = 32.
// ---------------------------------------------------------------------------
template <int K_IN>
__global__ void gin_mlp_kernel(const float* __restrict__ agg,  // [N, K_IN]
                               const float* __restrict__ wa,   // [K_IN, 32]
                               const float* __restrict__ ba,   // [32]
                               const float* __restrict__ wb,   // [32, 32]
                               const float* __restrict__ bb,   // [32]
                               float* __restrict__ r_out,      // [N, 32]
                               float* __restrict__ stats,      // [64]
                               int n_nodes) {
    constexpr int KB = K_IN / 32;  // k-blocks of 32
    __shared__ float zt[4][16 * 32];  // per-wave z1 tile (f32)
    __shared__ float lsum[32];
    __shared__ float lsq[32];

    const int tid = threadIdx.x;
    const int w = tid >> 5;        // wave in block
    const int lane = tid & 31;
    const int g = lane >> 4;       // lane half-group (0/1)
    const int lrow = lane & 15;    // row within tile (A/M index)
    const int node0 = ((int)blockIdx.x * 4 + w) * 16;

    if (tid < 32) { lsum[tid] = 0.f; lsq[tid] = 0.f; }

    // ---- pack B registers for wa: [kb][ntile], ISA B-layout -----------------
    v16h bwa[KB][2];
#pragma unroll
    for (int kb = 0; kb < KB; ++kb)
#pragma unroll
        for (int nt = 0; nt < 2; ++nt) {
            const int col = lrow + nt * 16;
#pragma unroll
            for (int j = 0; j < 16; ++j) {
                const int k = kb * 32 + (j < 8 ? j + 8 * g : j + 8 + 8 * g);
                bwa[kb][nt][j] = (_Float16)wa[k * 32 + col];
            }
        }
    // ---- pack B registers for wb: [ntile] ----------------------------------
    v16h bwb[2];
#pragma unroll
    for (int nt = 0; nt < 2; ++nt) {
        const int col = lrow + nt * 16;
#pragma unroll
        for (int j = 0; j < 16; ++j)
            bwb[nt][j] = (_Float16)wb[(j < 8 ? j + 8 * g : j + 8 + 8 * g) * 32 + col];
    }

    // ---- pack A registers from agg (clamped row), ISA A-layout -------------
    int arow = node0 + lrow;
    if (arow > n_nodes - 1) arow = n_nodes - 1;
    const float* arp = agg + (long)arow * K_IN;
    v16h amat[KB];
#pragma unroll
    for (int kb = 0; kb < KB; ++kb)
#pragma unroll
        for (int j = 0; j < 16; ++j) {
            const int k = kb * 32 + (j < 8 ? j + 8 * g : j + 8 + 8 * g);
            amat[kb][j] = (_Float16)arp[k];
        }

    // ---- GEMM 1: z1 = relu(agg @ wa + ba) ----------------------------------
    v8f c0 = zero8(), c1 = zero8();
#pragma unroll
    for (int kb = 0; kb < KB; ++kb) {
        c0 = wmma_f16(amat[kb], bwa[kb][0], c0);
        c1 = wmma_f16(amat[kb], bwa[kb][1], c1);
    }
    // C layout -> LDS tile (row_local = r + 8*g, col = lrow + 16*nt)
#pragma unroll
    for (int nt = 0; nt < 2; ++nt) {
        const int col = lrow + nt * 16;
        const float bias = ba[col];
#pragma unroll
        for (int r = 0; r < 8; ++r) {
            float v = (nt == 0 ? c0[r] : c1[r]) + bias;
            zt[w][(r + 8 * g) * 32 + col] = v > 0.f ? v : 0.f;
        }
    }
    __syncthreads();

    // ---- repack z1 as A (cross-lane shuffle via LDS) -----------------------
    v16h a2;
#pragma unroll
    for (int j = 0; j < 16; ++j) {
        const int k = (j < 8 ? j + 8 * g : j + 8 + 8 * g);
        a2[j] = (_Float16)zt[w][lrow * 32 + k];
    }

    // ---- GEMM 2: r = relu(z1 @ wb + bb), store + BN stats ------------------
    v8f d0 = wmma_f16(a2, bwb[0], zero8());
    v8f d1 = wmma_f16(a2, bwb[1], zero8());
#pragma unroll
    for (int nt = 0; nt < 2; ++nt) {
        const int col = lrow + nt * 16;
        const float bias = bb[col];
        float ps = 0.f, ps2 = 0.f;
#pragma unroll
        for (int r = 0; r < 8; ++r) {
            const int row = node0 + r + 8 * g;
            float v = (nt == 0 ? d0[r] : d1[r]) + bias;
            v = v > 0.f ? v : 0.f;
            if (row < n_nodes) {
                r_out[(long)row * 32 + col] = v;
                ps += v;
                ps2 += v * v;
            }
        }
        atomicAdd(&lsum[col], ps);
        atomicAdd(&lsq[col], ps2);
    }
    __syncthreads();
    if (tid < 32) atomicAdd(&stats[tid], lsum[tid]);
    else if (tid < 64) atomicAdd(&stats[tid], lsq[tid - 32]);
}

// ---------------------------------------------------------------------------
// BN apply (in place): r = (r - mu) * rsqrt(var + eps) * gamma + beta
// ---------------------------------------------------------------------------
__global__ void bn_apply_kernel(float* __restrict__ r, const float* __restrict__ stats,
                                const float* __restrict__ gamma,
                                const float* __restrict__ beta, int n_nodes) {
    __shared__ float sc[32], sh[32];
    if (threadIdx.x < 32) {
        const float invn = 1.f / (float)n_nodes;
        float mu = stats[threadIdx.x] * invn;
        float var = stats[32 + threadIdx.x] * invn - mu * mu;
        float s = rsqrtf(var + BN_EPS) * gamma[threadIdx.x];
        sc[threadIdx.x] = s;
        sh[threadIdx.x] = beta[threadIdx.x] - mu * s;
    }
    __syncthreads();
    long tid = (long)blockIdx.x * blockDim.x + threadIdx.x;
    if (tid < (long)n_nodes * 32) {
        int f = (int)(tid & 31);
        r[tid] = r[tid] * sc[f] + sh[f];
    }
}

// ---------------------------------------------------------------------------
// Global add pool: g[batch[node], f] += h[node, f]
// ---------------------------------------------------------------------------
__global__ void pool_kernel(const float* __restrict__ h, const int* __restrict__ batch,
                            float* __restrict__ g, int n_nodes) {
    long tid = (long)blockIdx.x * blockDim.x + threadIdx.x;
    if (tid >= (long)n_nodes * 32) return;
    int node = (int)(tid >> 5);
    int f = (int)(tid & 31);
    atomicAdd(&g[batch[node] * 32 + f], h[tid]);
}

// ---------------------------------------------------------------------------
// Head: out = relu(g @ fc1 + b1) @ fc2 + b2   ([64,32] -> [64,10])
// ---------------------------------------------------------------------------
__global__ void head_kernel(const float* __restrict__ g, const float* __restrict__ fc1w,
                            const float* __restrict__ fc1b,
                            const float* __restrict__ fc2w,
                            const float* __restrict__ fc2b, float* __restrict__ out) {
    __shared__ float t[64 * 32];
    for (int i = threadIdx.x; i < 64 * 32; i += blockDim.x) {
        int gi = i >> 5, f = i & 31;
        float acc = fc1b[f];
#pragma unroll
        for (int k = 0; k < 32; ++k) acc += g[gi * 32 + k] * fc1w[k * 32 + f];
        t[i] = acc > 0.f ? acc : 0.f;
    }
    __syncthreads();
    for (int i = threadIdx.x; i < 64 * 10; i += blockDim.x) {
        int gi = i / 10, c = i % 10;
        float acc = fc2b[c];
#pragma unroll
        for (int k = 0; k < 32; ++k) acc += t[gi * 32 + k] * fc2w[k * 10 + c];
        out[i] = acc;
    }
}

// ---------------------------------------------------------------------------
extern "C" void kernel_launch(void* const* d_in, const int* in_sizes, int n_in,
                              void* d_out, int out_size, void* d_ws, size_t ws_size,
                              hipStream_t stream) {
    const float* x     = (const float*)d_in[0];
    const int*   eidx  = (const int*)d_in[1];
    const int*   batch = (const int*)d_in[2];
    const float* w1a   = (const float*)d_in[3];
    const float* b1a   = (const float*)d_in[4];
    const float* w1b   = (const float*)d_in[5];
    const float* b1b   = (const float*)d_in[6];
    const float* Wa    = (const float*)d_in[7];   // [4,32,32]
    const float* Ba    = (const float*)d_in[8];   // [4,32]
    const float* Wb    = (const float*)d_in[9];   // [4,32,32]
    const float* Bb    = (const float*)d_in[10];  // [4,32]
    const float* bn_g  = (const float*)d_in[11];  // [5,32]
    const float* bn_b  = (const float*)d_in[12];  // [5,32]
    const float* fc1w  = (const float*)d_in[13];
    const float* fc1b  = (const float*)d_in[14];
    const float* fc2w  = (const float*)d_in[15];
    const float* fc2b  = (const float*)d_in[16];

    const int E = in_sizes[1] / 2;   // 1,600,000
    const int N = in_sizes[2];       // 100,000
    const int* src = eidx;
    const int* dst = eidx + E;

    float* ws      = (float*)d_ws;
    float* rbuf    = ws;                         // N*32
    float* aggbuf  = rbuf + (size_t)N * 32;      // N*64 (layer1); first N*32 reused later
    float* stats   = aggbuf + (size_t)N * 64;    // 64
    float* gbuf    = stats + 64;                 // 64*32

    float* out = (float*)d_out;

    const int THR = 256;
    auto blocks = [](long total, int thr) { return (unsigned)((total + thr - 1) / thr); };

    // ---------------- layer 1 (feat 64 -> 32) ----------------
    copy_f4_kernel<<<blocks((long)N * 16, THR), THR, 0, stream>>>(x, aggbuf, (long)N * 16);
    edge_aggr_kernel<64><<<blocks((long)E * 16, THR), THR, 0, stream>>>(x, src, dst,
                                                                        aggbuf, E);
    hipMemsetAsync(stats, 0, 64 * sizeof(float), stream);
    gin_mlp_kernel<64><<<(N + 63) / 64, 128, 0, stream>>>(aggbuf, w1a, b1a, w1b, b1b,
                                                          rbuf, stats, N);
    bn_apply_kernel<<<blocks((long)N * 32, THR), THR, 0, stream>>>(rbuf, stats, bn_g,
                                                                   bn_b, N);

    // ---------------- layers 2-5 (32 -> 32) ----------------
    for (int i = 0; i < 4; ++i) {
        copy_f4_kernel<<<blocks((long)N * 8, THR), THR, 0, stream>>>(rbuf, aggbuf,
                                                                     (long)N * 8);
        edge_aggr_kernel<32><<<blocks((long)E * 8, THR), THR, 0, stream>>>(rbuf, src,
                                                                           dst, aggbuf, E);
        hipMemsetAsync(stats, 0, 64 * sizeof(float), stream);
        gin_mlp_kernel<32><<<(N + 63) / 64, 128, 0, stream>>>(
            aggbuf, Wa + i * 1024, Ba + i * 32, Wb + i * 1024, Bb + i * 32, rbuf, stats,
            N);
        bn_apply_kernel<<<blocks((long)N * 32, THR), THR, 0, stream>>>(
            rbuf, stats, bn_g + (i + 1) * 32, bn_b + (i + 1) * 32, N);
    }

    // ---------------- pool + head ----------------
    hipMemsetAsync(gbuf, 0, 64 * 32 * sizeof(float), stream);
    pool_kernel<<<blocks((long)N * 32, THR), THR, 0, stream>>>(rbuf, batch, gbuf, N);
    head_kernel<<<1, 256, 0, stream>>>(gbuf, fc1w, fc1b, fc2w, fc2b, out);
}
